// MoePreMix_37254546325634
// MI455X (gfx1250) — matrix operand order repaired and checked by reading, not compiled, not run on hardware
//
#include <hip/hip_runtime.h>
#include <hip/hip_bf16.h>

// ---------------- problem constants ----------------
#define BB 4
#define SS 2048
#define EE 1024
#define HH 16
#define PP 8
#define DD 64
#define NTOK (BB * SS)              // 8192 tokens
#define QSIZE (BB * HH * SS * DD)   // 8388608 floats per tensor (q, k, v)
#define NQKW (2 * EE * EE)          // 2097152 qk_w elements

// SqSoftplus constants
#define SQ_B 2.0480546988460353f
#define SQ_C 0.841470984807896f
#define SQ_A 0.45339765151640377f
#define SQ_INVB 0.48826828005937f   // 1/SQ_B
#define SQ_INVC 1.18840346918157f   // 1/SQ_C
#define NU_C 0.9999993273473248f    // log(2.71828)
#define EPS_C 1e-6f

typedef __attribute__((ext_vector_type(16))) _Float16 v16h;
typedef __attribute__((ext_vector_type(8)))  _Float16 v8h;
typedef __attribute__((ext_vector_type(8)))  float    v8f;

// ---------------- math helpers ----------------
__device__ __forceinline__ float softplus_f(float y) {
    return fmaxf(y, 0.f) + __logf(1.f + __expf(-fabsf(y)));
}

__device__ __forceinline__ float sqsp_f(float v) {
    float t = (1.f - SQ_A) * (softplus_f(SQ_B * v) * SQ_INVB)
            + SQ_A * (softplus_f(SQ_C * v) * SQ_INVC);
    return t * t;
}

__device__ __forceinline__ float wave_reduce_add(float v) {
#pragma unroll
    for (int m = 16; m > 0; m >>= 1) v += __shfl_xor(v, m, 32);
    return v;
}

// ---------------- kernel 0a: Wsum[h,p,o] = sum_j W[h,p,o,j] ---------------
__global__ void wsum_kernel(const float* __restrict__ W1,
                            const float* __restrict__ W2,
                            float* __restrict__ ws) {
    int idx = blockIdx.x * blockDim.x + threadIdx.x;   // 0..16383
    const float* W = (idx < HH * PP * DD) ? W1 : W2;
    int o = idx & (HH * PP * DD - 1);
    const float4* p = (const float4*)(W + (size_t)o * DD);
    float s = 0.f;
#pragma unroll
    for (int i = 0; i < DD / 4; ++i) {
        float4 t = p[i];
        s += t.x + t.y + t.z + t.w;
    }
    ws[idx] = s;
}

// ---------------- kernel 0b: split qk_w into f16 hi/lo planes -------------
__global__ void split_qkw_kernel(const float* __restrict__ qkw,
                                 _Float16* __restrict__ bh,
                                 _Float16* __restrict__ bl) {
    int i = (blockIdx.x * blockDim.x + threadIdx.x) * 8;
    float4 t0 = *(const float4*)(qkw + i);
    float4 t1 = *(const float4*)(qkw + i + 4);
    float av[8] = {t0.x, t0.y, t0.z, t0.w, t1.x, t1.y, t1.z, t1.w};
    v8h hv, lv;
#pragma unroll
    for (int j = 0; j < 8; ++j) {
        _Float16 hh = (_Float16)av[j];
        hv[j] = hh;
        lv[j] = (_Float16)(av[j] - (float)hh);
    }
    *(v8h*)(bh + i) = hv;
    *(v8h*)(bl + i) = lv;
}

// ---------------- GEMM building blocks ----------------
struct ARaw { float4 t0, t1, t2, t3; };

__device__ __forceinline__ ARaw load_a(const float* arow, int k, int q) {
    ARaw r;
    const float4* p0 = (const float4*)(arow + k + 16 * q);
    const float4* p1 = (const float4*)(arow + k + 16 * q + 8);
    r.t0 = p0[0]; r.t1 = p0[1];
    r.t2 = p1[0]; r.t3 = p1[1];
    return r;
}

__device__ __forceinline__ void split_a(const ARaw& r, v16h& ah, v16h& al) {
    float av[16] = {r.t0.x, r.t0.y, r.t0.z, r.t0.w, r.t1.x, r.t1.y, r.t1.z, r.t1.w,
                    r.t2.x, r.t2.y, r.t2.z, r.t2.w, r.t3.x, r.t3.y, r.t3.z, r.t3.w};
#pragma unroll
    for (int i = 0; i < 16; ++i) {
        _Float16 hh = (_Float16)av[i];
        ah[i] = hh;
        al[i] = (_Float16)(av[i] - (float)hh);
    }
}

__device__ __forceinline__ void load_bfrags(const _Float16* const* bhp,
                                            const _Float16* const* blp,
                                            int k, v16h* Bh, v16h* Bl) {
#pragma unroll
    for (int j = 0; j < 4; ++j) {
        Bh[j] = *(const v16h*)(bhp[j] + k);
        Bl[j] = *(const v16h*)(blp[j] + k);
    }
}

__device__ __forceinline__ void wmma12(const v16h& ah, const v16h& al,
                                       const v16h* Bh, const v16h* Bl, v8f* c) {
    // term-major: dependent WMMAs on the same accumulator are 4 apart
#pragma unroll
    for (int j = 0; j < 4; ++j)
        c[j] = __builtin_amdgcn_wmma_f32_16x16x32_f16(false, ah, false, Bh[j],
                                                      (short)0, c[j], false, false);
#pragma unroll
    for (int j = 0; j < 4; ++j)
        c[j] = __builtin_amdgcn_wmma_f32_16x16x32_f16(false, al, false, Bh[j],
                                                      (short)0, c[j], false, false);
#pragma unroll
    for (int j = 0; j < 4; ++j)
        c[j] = __builtin_amdgcn_wmma_f32_16x16x32_f16(false, ah, false, Bl[j],
                                                      (short)0, c[j], false, false);
}

// ---------------- kernel 1: qk = x @ qk_w.T via split-f16 WMMA ------------
// M=8192, N=2048, K=1024.  Per K=32 step: 12 x V_WMMA_F32_16X16X32_F16
// (4 tiles x {hi*hi, lo*hi, hi*lo}).  Register double-buffering: step k+32
// loads are issued before step k's WMMAs so waits are partial and the four
// independent accumulator chains fill the WMMA->WMMA hazard window.
__global__ void __launch_bounds__(256, 1)
qk_wmma_kernel(const float* __restrict__ x,
               const _Float16* __restrict__ bhp_g,
               const _Float16* __restrict__ blp_g,
               float* __restrict__ out) {
    const int lane = threadIdx.x & 31;
    const int wave = threadIdx.x >> 5;
    const int bm = blockIdx.x & 63;     // 64 M-blocks of 128
    const int bn = blockIdx.x >> 6;     // 32 N-blocks of 64
    const int m0 = bm * 128 + wave * 16;
    const int n0 = bn * 64;

    const int l15 = lane & 15;
    const int q = lane >> 4;            // lane-half selects K-half (contiguous k = 16q + s, same for A and B)

    const float* arow = x + (size_t)(m0 + l15) * EE;

    const _Float16* bhp[4];
    const _Float16* blp[4];
#pragma unroll
    for (int j = 0; j < 4; ++j) {
        size_t off = (size_t)(n0 + j * 16 + l15) * EE + 16 * q;
        bhp[j] = bhp_g + off;
        blp[j] = blp_g + off;
    }

    v8f c[4] = {{}, {}, {}, {}};

    v16h Bh0[4], Bl0[4], Bh1[4], Bl1[4];
    ARaw A0, A1;

    // prologue: stage step k=0
    load_bfrags(bhp, blp, 0, Bh0, Bl0);
    A0 = load_a(arow, 0, q);

    for (int k = 0; k < EE; k += 64) {
        const int k1 = k + 32;
        const int k2 = (k + 64) & (EE - 1);   // final prefetch wraps (safe, discarded)

        // prefetch step k+32 into buffer 1
        load_bfrags(bhp, blp, k1, Bh1, Bl1);
        A1 = load_a(arow, k1, q);

        // compute step k from buffer 0
        v16h ah, al;
        split_a(A0, ah, al);
        wmma12(ah, al, Bh0, Bl0, c);

        // prefetch step k+64 into buffer 0
        load_bfrags(bhp, blp, k2, Bh0, Bl0);
        A0 = load_a(arow, k2, q);

        // compute step k+32 from buffer 1
        split_a(A1, ah, al);
        wmma12(ah, al, Bh1, Bl1, c);
    }

    // Write-out: C/D layout: VGPR v, lanes 0-15 -> M=v, lanes 16-31 -> M=v+8.
#pragma unroll
    for (int j = 0; j < 4; ++j) {
        int col = n0 + j * 16 + l15;
        int base;
        float* dst;
        if (col < EE) {
            dst = out;
            base = col;
        } else {
            dst = out + QSIZE;
            base = col - EE;
        }
        int h = base >> 6;
        int d = base & 63;
#pragma unroll
        for (int v = 0; v < 8; ++v) {
            int m = m0 + v + 8 * q;
            int b = m >> 11;
            int s = m & (SS - 1);
            dst[(((size_t)(b * HH + h) * SS + s) * DD) + d] = c[j][v];
        }
    }
}

// ---------------- kernel 2: vector hull (one wave per (head, token)) ------
__global__ void vhull_kernel(const float* __restrict__ x,
                             const float* __restrict__ ws,   // [W1s | W2s]
                             const float* __restrict__ b1,
                             const float* __restrict__ b2,
                             const float* __restrict__ gate_w,
                             const float* __restrict__ gate_b,
                             float* __restrict__ vout) {
    const int lane = threadIdx.x & 31;
    const int wid = blockIdx.x * (blockDim.x >> 5) + (threadIdx.x >> 5);
    const int h = wid >> 13;
    const int t = wid & (NTOK - 1);

    const float* xrow = x + (size_t)t * EE + h * DD;
    const float* gwp = gate_w + h * DD;
    const float* W1s = ws;
    const float* W2s = ws + HH * PP * DD;

    const int d0 = lane, d1 = lane + 32;
    float x0 = xrow[d0], x1 = xrow[d1];
    float w0 = gwp[d0],  w1 = gwp[d1];

    float dot = wave_reduce_add(fmaf(x0, w0, x1 * w1)) + gate_b[h];
    float u = sqsp_f(dot);
    float g = 1.f - __expf(-u);
    float xg0 = x0 * g, xg1 = x1 * g;

    float msq = wave_reduce_add(fmaf(xg0, xg0, xg1 * xg1)) * (1.f / DD);
    float tau = sqrtf(msq + EPS_C + NU_C);

    float s1 = wave_reduce_add(xg0 + xg1);

    float sc0[PP], sc1[PP];
#pragma unroll
    for (int p = 0; p < PP; ++p) {
        int off = (h * PP + p) * DD;
        float a0 = sqsp_f(fmaf(s1, W1s[off + d0], b1[off + d0]));
        float a1 = sqsp_f(fmaf(s1, W1s[off + d1], b1[off + d1]));
        float s2 = wave_reduce_add(a0 + a1);
        float y0 = sqsp_f(fmaf(s2, W2s[off + d0], b2[off + d0]));
        float y1 = sqsp_f(fmaf(s2, W2s[off + d1], b2[off + d1]));
        sc0[p] = y0 * tau;
        sc1[p] = y1 * tau;
    }

    float m0 = sc0[0], m1 = sc1[0];
#pragma unroll
    for (int p = 1; p < PP; ++p) { m0 = fmaxf(m0, sc0[p]); m1 = fmaxf(m1, sc1[p]); }
    float e0 = 0.f, e1 = 0.f;
#pragma unroll
    for (int p = 0; p < PP; ++p) { e0 += __expf(sc0[p] - m0); e1 += __expf(sc1[p] - m1); }
    float inv_tau = 1.f / tau;
    float v0 = (m0 + __logf(e0)) * inv_tau;
    float v1 = (m1 + __logf(e1)) * inv_tau;

    int b = t >> 11;
    int s = t & (SS - 1);
    float* dst = vout + ((size_t)(b * HH + h) * SS + s) * DD;
    dst[d0] = v0;
    dst[d1] = v1;
}

// ---------------- launch ----------------
extern "C" void kernel_launch(void* const* d_in, const int* in_sizes, int n_in,
                              void* d_out, int out_size, void* d_ws, size_t ws_size,
                              hipStream_t stream) {
    const float* x    = (const float*)d_in[0];
    const float* qkw  = (const float*)d_in[1];
    const float* W1   = (const float*)d_in[2];
    const float* b1   = (const float*)d_in[3];
    const float* W2   = (const float*)d_in[4];
    const float* b2   = (const float*)d_in[5];
    const float* gw   = (const float*)d_in[6];
    const float* gb   = (const float*)d_in[7];
    float* out = (float*)d_out;
    float* ws  = (float*)d_ws;

    // workspace layout: [Wsum: 16384 f32][Bhi: NQKW f16][Blo: NQKW f16]
    _Float16* bh = (_Float16*)(ws + 16384);
    _Float16* bl = bh + NQKW;

    // 0a) weight-sum reduction
    wsum_kernel<<<64, 256, 0, stream>>>(W1, W2, ws);

    // 0b) split qk_w into f16 hi/lo planes (8 MB, one pass)
    split_qkw_kernel<<<NQKW / (256 * 8), 256, 0, stream>>>(qkw, bh, bl);

    // 1) q,k projection GEMM: 2048 blocks x 256 threads
    qk_wmma_kernel<<<64 * 32, 256, 0, stream>>>(x, bh, bl, out);

    // 2) v hull: one wave per (head, token)
    vhull_kernel<<<(HH * NTOK) / 8, 256, 0, stream>>>(x, ws, b1, b2, gw, gb,
                                                      out + 2 * (size_t)QSIZE);
}
